// MoERouter_27324581937467
// MI455X (gfx1250) — matrix operand order repaired and checked by reading, not compiled, not run on hardware
//
#include <hip/hip_runtime.h>
#include <hip/hip_bf16.h>

// ---------------------------------------------------------------------------
// MoE router for MI455X (gfx1250, wave32).
//   logits = x @ W^T + b          (bf16x3 split-precision WMMA GEMM)
//   softmax -> top-8 -> normalized weights, indices, one-hot mask
// ---------------------------------------------------------------------------

typedef __attribute__((ext_vector_type(16))) __bf16 v16bf;
typedef __attribute__((ext_vector_type(8)))  float  v8f;
typedef __attribute__((ext_vector_type(4)))  float  v4f;

#define HIDDEN      4096
#define NUM_EXPERTS 64
#define TOP_K       8
#define N_TOKENS    16384

// ---------------------------------------------------------------------------
// Kernel 1: split W (fp32) into bf16 hi/lo parts once.  W is 1 MB total in
// bf16 form -> lives in the 192 MB L2 for the whole GEMM.
// ---------------------------------------------------------------------------
__global__ __launch_bounds__(256) void conv_w_kernel(
    const float* __restrict__ w, __bf16* __restrict__ whi,
    __bf16* __restrict__ wlo)
{
    int i = blockIdx.x * 256 + threadIdx.x;     // E*H = 262144 elements
    float f = w[i];
    __bf16 hi = (__bf16)f;                      // RNE round to bf16
    whi[i] = hi;
    wlo[i] = (__bf16)(f - (float)hi);           // residual
}

// ---------------------------------------------------------------------------
// Kernel 2: GEMM logits[N,64] = x[N,4096] @ W^T + b, bf16x3 WMMA.
// Block = 256 threads = 8 waves; each wave computes a 16-token x 64-expert
// strip (four 16x16 f32 accumulators).  K-chunk = 32 (one bf16 WMMA depth).
//
// Fragment layouts (wave32; C/D per CDNA5 ISA 7.12.2, A/B use this code's own contiguous k order):
//   A 16x32 bf16 : lane l (m=l&15, h=l>>4) holds K = [16h .. 16h+15]
//   B 32x16 bf16 : lane l (n=l&15, h=l>>4) holds K = [16h .. 16h+15]
//   C/D 16x16 f32: lane l (n=l&15), VGPR v holds M = v + 8*(l>>4)
// ---------------------------------------------------------------------------
__global__ __launch_bounds__(256) void moe_gate_gemm(
    const float*  __restrict__ x,
    const __bf16* __restrict__ whi,
    const __bf16* __restrict__ wlo,
    const float*  __restrict__ bias,
    float*        __restrict__ logits)
{
    const int lane  = threadIdx.x & 31;
    const int wave  = threadIdx.x >> 5;
    const int strip = blockIdx.x * 8 + wave;    // 16-token strip index
    const int m     = lane & 15;
    const int h     = lane >> 4;

    const float* xrow = x + (size_t)(strip * 16 + m) * HIDDEN;

    v8f acc[4] = {};                            // 4 expert tiles of 16

    for (int k0 = 0; k0 < HIDDEN; k0 += 32) {
        // ---- A fragment: load 16 fp32, split into bf16 hi/lo in registers
        float xv[16];
        *(v4f*)(xv + 0)  = *(const v4f*)(xrow + k0 + 16 * h);
        *(v4f*)(xv + 4)  = *(const v4f*)(xrow + k0 + 16 * h + 4);
        *(v4f*)(xv + 8)  = *(const v4f*)(xrow + k0 + 16 * h + 8);
        *(v4f*)(xv + 12) = *(const v4f*)(xrow + k0 + 16 * h + 12);

        v16bf ahi, alo;
#pragma unroll
        for (int i = 0; i < 16; ++i) {
            __bf16 hi = (__bf16)xv[i];
            ahi[i] = hi;
            alo[i] = (__bf16)(xv[i] - (float)hi);
        }

        // ---- B fragments (already bf16 in scratch, L2-resident) + WMMA
#pragma unroll
        for (int t = 0; t < 4; ++t) {
            const size_t boff = (size_t)(t * 16 + m) * HIDDEN + k0 + 16 * h;
            v16bf bhi = *(const v16bf*)(whi + boff);
            v16bf blo = *(const v16bf*)(wlo + boff);

            acc[t] = __builtin_amdgcn_wmma_f32_16x16x32_bf16(
                false, ahi, false, bhi, (short)0, acc[t], false, false);
            acc[t] = __builtin_amdgcn_wmma_f32_16x16x32_bf16(
                false, ahi, false, blo, (short)0, acc[t], false, false);
            acc[t] = __builtin_amdgcn_wmma_f32_16x16x32_bf16(
                false, alo, false, bhi, (short)0, acc[t], false, false);
        }
    }

    // ---- epilogue: add bias, store logits [N, 64]
#pragma unroll
    for (int t = 0; t < 4; ++t) {
        const float b = bias[t * 16 + m];       // expert column bias
        float* outp = logits + (size_t)(strip * 16) * NUM_EXPERTS + t * 16 + m;
#pragma unroll
        for (int v = 0; v < 8; ++v) {
            const int rowm = v + 8 * h;
            outp[(size_t)rowm * NUM_EXPERTS] = acc[t][v] + b;
        }
    }
}

// ---------------------------------------------------------------------------
// Kernel 3: zero-fill the one-hot mask region (poisoned by harness).
// ---------------------------------------------------------------------------
__global__ __launch_bounds__(256) void fill_zero_kernel(float4* __restrict__ p)
{
    p[(size_t)blockIdx.x * 256 + threadIdx.x] = make_float4(0.f, 0.f, 0.f, 0.f);
}

// ---------------------------------------------------------------------------
// Kernel 4: softmax + top-8 per token, one wave32 per token.
//   lane holds experts {lane, lane+32}; shuffle-xor tree reductions.
//   Tie-break: lower expert index wins (matches jax.lax.top_k).
// ---------------------------------------------------------------------------
__global__ __launch_bounds__(256) void moe_topk_kernel(
    const float* __restrict__ logits,
    float* __restrict__ outw, float* __restrict__ outi,
    float* __restrict__ mask)
{
    const int lane = threadIdx.x & 31;
    const int wave = threadIdx.x >> 5;
    const int n    = blockIdx.x * 8 + wave;

    const float* lrow = logits + (size_t)n * NUM_EXPERTS;
    float l0 = lrow[lane];
    float l1 = lrow[lane + 32];

    // softmax over 64 experts
    float mx = fmaxf(l0, l1);
#pragma unroll
    for (int off = 16; off; off >>= 1) mx = fmaxf(mx, __shfl_xor(mx, off, 32));
    float e0 = __expf(l0 - mx), e1 = __expf(l1 - mx);
    float s = e0 + e1;
#pragma unroll
    for (int off = 16; off; off >>= 1) s += __shfl_xor(s, off, 32);
    const float inv = 1.0f / s;
    float p0 = e0 * inv, p1 = e1 * inv;

    // iterative top-8 arg-max with lower-index tie break
    float myv = 0.f;
    int   myi = 0;
    float ksum = 0.f;
#pragma unroll
    for (int kk = 0; kk < TOP_K; ++kk) {
        float v = p0; int i = lane;
        if (p1 > v) { v = p1; i = lane + 32; }
#pragma unroll
        for (int off = 16; off; off >>= 1) {
            float ov = __shfl_xor(v, off, 32);
            int   oi = __shfl_xor(i, off, 32);
            if (ov > v || (ov == v && oi < i)) { v = ov; i = oi; }
        }
        ksum += v;                          // same on all lanes
        if (lane == kk) { myv = v; myi = i; }
        if (i == lane)      p0 = -1.0f;     // knock out selected expert
        if (i == lane + 32) p1 = -1.0f;
    }

    if (lane < TOP_K) {
        outw[(size_t)n * TOP_K + lane] = myv / ksum;         // renormalized
        outi[(size_t)n * TOP_K + lane] = (float)myi;         // expert index
        // expert_mask[E][K][N] one-hot scatter
        mask[((size_t)myi * TOP_K + lane) * N_TOKENS + n] = 1.0f;
    }
}

// ---------------------------------------------------------------------------
extern "C" void kernel_launch(void* const* d_in, const int* in_sizes, int n_in,
                              void* d_out, int out_size, void* d_ws, size_t ws_size,
                              hipStream_t stream)
{
    const float* x      = (const float*)d_in[0];   // [N, H]
    const float* gate_w = (const float*)d_in[1];   // [E, H]
    const float* gate_b = (const float*)d_in[2];   // [E]

    float* logits = (float*)d_out;                                  // N*E
    float* outw   = logits + (size_t)N_TOKENS * NUM_EXPERTS;        // N*K
    float* outi   = outw + (size_t)N_TOKENS * TOP_K;                // N*K
    float* mask   = outi + (size_t)N_TOKENS * TOP_K;                // E*K*N

    __bf16* whi = (__bf16*)d_ws;                                    // 512 KB
    __bf16* wlo = whi + (size_t)NUM_EXPERTS * HIDDEN;               // 512 KB

    // 1) split W into bf16 hi/lo (262144 elements)
    conv_w_kernel<<<(NUM_EXPERTS * HIDDEN) / 256, 256, 0, stream>>>(
        gate_w, whi, wlo);

    // 2) WMMA GEMM: 128 blocks x 8 waves, each wave = 16 tokens x 64 experts
    moe_gate_gemm<<<N_TOKENS / 128, 256, 0, stream>>>(
        x, whi, wlo, gate_b, logits);

    // 3) zero the one-hot mask (E*K*N = 8388608 floats = 2097152 float4)
    const size_t mask_f4 = (size_t)NUM_EXPERTS * TOP_K * N_TOKENS / 4;
    fill_zero_kernel<<<(unsigned)(mask_f4 / 256), 256, 0, stream>>>(
        (float4*)mask);

    // 4) softmax + top-8 (+ scatter ones into mask): one wave per token
    moe_topk_kernel<<<N_TOKENS / 8, 256, 0, stream>>>(
        logits, outw, outi, mask);
}
